// GraphTransformerEncoder_45861660787164
// MI455X (gfx1250) — compile-verified, matrix operand order checked by reading, not run on hardware
//
#include <hip/hip_runtime.h>
#include <hip/hip_bf16.h>
#include <math.h>

// ---- problem constants ----
#define Bn   16
#define Nn   512
#define Dm   256
#define Hh   8
#define DK   32
#define Ll   6
#define DFFm 1024
#define PEm  64
#define Em   131072
#define BNm  (Bn*Nn)           // 8192

typedef __bf16 bf16_t;
typedef __attribute__((ext_vector_type(16))) __bf16 v16bf;
typedef __attribute__((ext_vector_type(8)))  __bf16 v8bf;
typedef __attribute__((ext_vector_type(8)))  float  v8f;

// Build a 16-element bf16 A/B fragment from two contiguous 8-element (16B) loads.
__device__ inline v16bf load_frag(const bf16_t* p0, const bf16_t* p1) {
    v8bf a = *(const v8bf*)p0;
    v8bf b = *(const v8bf*)p1;
    v16bf r;
#pragma unroll
    for (int i = 0; i < 8; ++i) { r[i] = a[i]; r[8 + i] = b[i]; }
    return r;
}

__device__ inline v8f wmma_bf16(v16bf a, v16bf b, v8f c) {
    return __builtin_amdgcn_wmma_f32_16x16x32_bf16(false, a, false, b, (short)0, c, false, false);
}

// ---------------------------------------------------------------------------
// 0) f32 -> bf16 transpose-convert for weights: in (L,K,Nc) -> out (L,Nc,K)
// ---------------------------------------------------------------------------
__global__ void conv_t_kernel(const float* __restrict__ in, bf16_t* __restrict__ out,
                              int Lb, int K, int Nc) {
    size_t idx = (size_t)blockIdx.x * blockDim.x + threadIdx.x;
    size_t per = (size_t)K * Nc;
    size_t tot = (size_t)Lb * per;
    if (idx >= tot) return;
    size_t l = idx / per;
    size_t rkn = idx % per;
    size_t n = rkn / K;
    size_t k = rkn % K;
    out[idx] = (bf16_t)in[l * per + k * Nc + n];
}

// ---------------------------------------------------------------------------
// 1) edge scalar constants: wbar = mean(edge_W), bbar = mean(edge_b)
// ---------------------------------------------------------------------------
__global__ void edge_const_kernel(const float* __restrict__ eW, const float* __restrict__ eB,
                                  float* __restrict__ consts) {
    __shared__ float rw[Dm], rb[Dm];
    int t = threadIdx.x;
    rw[t] = eW[t]; rb[t] = eB[t];
    __syncthreads();
    for (int s = Dm / 2; s > 0; s >>= 1) {
        if (t < s) { rw[t] += rw[t + s]; rb[t] += rb[t + s]; }
        __syncthreads();
    }
    if (t == 0) { consts[0] = rw[0] * (1.f / Dm); consts[1] = rb[0] * (1.f / Dm); }
}

__global__ void fill_zero_kernel(float* __restrict__ p, size_t n) {
    size_t i = (size_t)blockIdx.x * blockDim.x + threadIdx.x;
    if (i < n) p[i] = 0.f;
}

__global__ void edge_scatter_kernel(const int* __restrict__ eidx, const float* __restrict__ eattr,
                                    const float* __restrict__ consts, float* __restrict__ ew) {
    int e = blockIdx.x * blockDim.x + threadIdx.x;
    if (e >= Em) return;
    int b = e >> 13;                       // / 8192 edges per graph
    int src = eidx[e]        - b * Nn;
    int dst = eidx[Em + e]   - b * Nn;
    ew[((size_t)b * Nn + src) * Nn + dst] = eattr[e] * consts[0] + consts[1];
}

// ---------------------------------------------------------------------------
// 2) embedding: h = x@node_W + node_b + pe@pe_W + pe_b ; writes f32 + bf16
// ---------------------------------------------------------------------------
__global__ void embed_kernel(const float* __restrict__ x,
                             const float* __restrict__ nodeW, const float* __restrict__ nodeB,
                             const float* __restrict__ peW,  const float* __restrict__ peB,
                             float* __restrict__ hF, bf16_t* __restrict__ hB) {
    __shared__ float pe[PEm];
    int n = blockIdx.x;
    int t = threadIdx.x;
    float x0 = x[n * 3 + 0], x1 = x[n * 3 + 1], x2 = x[n * 3 + 2];
    if (t < PEm) {
        int i = t >> 2, ph = t & 3;
        // dt_i = 10000^(-i/16)
        float dt = expf(-(float)i * (9.210340371976184f / 16.f));
        float cx = x0 * 0.01f, cy = x1 * 0.01f;
        float arg = ((ph < 2) ? cx : cy) * dt;
        pe[t] = (ph & 1) ? cosf(arg) : sinf(arg);
    }
    __syncthreads();
    int d = t;
    float acc = nodeB[d] + peB[d]
              + x0 * nodeW[d] + x1 * nodeW[Dm + d] + x2 * nodeW[2 * Dm + d];
#pragma unroll 8
    for (int k = 0; k < PEm; ++k) acc += pe[k] * peW[k * Dm + d];
    size_t o = (size_t)n * Dm + d;
    hF[o] = acc;
    hB[o] = (bf16_t)acc;
}

// ---------------------------------------------------------------------------
// 3) WMMA GEMM: C[M,N] = A[M,K](bf16) x Bt[N,K](bf16, pre-transposed) + bias
//    Each wave computes a 32x32 output (2x2 tiles of 16x16) for 2x fragment
//    reuse out of L2. Block = 8 waves covering 64 rows x 128 cols.
// ---------------------------------------------------------------------------
__global__ void gemm_wmma_kernel(const bf16_t* __restrict__ A, const bf16_t* __restrict__ Bt,
                                 const float* __restrict__ bias,
                                 float* __restrict__ Cf, bf16_t* __restrict__ Cb,
                                 int M, int N, int K, int relu) {
    const int lane = threadIdx.x & 31;
    const int wave = threadIdx.x >> 5;
    const int wr = wave & 1;             // 2 row-waves
    const int wc = wave >> 1;            // 4 col-waves
    const int row0 = blockIdx.y * 64 + wr * 32;
    const int col0 = blockIdx.x * 128 + wc * 32;
    const int r  = lane & 15;            // A row / B column within tile
    const int kb = (lane >> 4) * 8;      // K-half select
    const int m0 = kb;                   // C/D row base

    const bf16_t* arow0 = A  + (size_t)(row0 + r) * K;
    const bf16_t* arow1 = A  + (size_t)(row0 + 16 + r) * K;
    const bf16_t* brow0 = Bt + (size_t)(col0 + r) * K;
    const bf16_t* brow1 = Bt + (size_t)(col0 + 16 + r) * K;

    v8f acc00 = {}, acc01 = {}, acc10 = {}, acc11 = {};
    for (int kk = 0; kk < K; kk += 32) {
        v16bf a0 = load_frag(arow0 + kk + kb, arow0 + kk + kb + 16);
        v16bf a1 = load_frag(arow1 + kk + kb, arow1 + kk + kb + 16);
        v16bf b0 = load_frag(brow0 + kk + kb, brow0 + kk + kb + 16);
        v16bf b1 = load_frag(brow1 + kk + kb, brow1 + kk + kb + 16);
        acc00 = wmma_bf16(a0, b0, acc00);
        acc01 = wmma_bf16(a0, b1, acc01);
        acc10 = wmma_bf16(a1, b0, acc10);
        acc11 = wmma_bf16(a1, b1, acc11);
    }
    float bv0 = bias ? bias[col0 + r]      : 0.f;
    float bv1 = bias ? bias[col0 + 16 + r] : 0.f;
#pragma unroll
    for (int j = 0; j < 8; ++j) {
        float v00 = acc00[j] + bv0;
        float v01 = acc01[j] + bv1;
        float v10 = acc10[j] + bv0;
        float v11 = acc11[j] + bv1;
        if (relu) {
            v00 = fmaxf(v00, 0.f); v01 = fmaxf(v01, 0.f);
            v10 = fmaxf(v10, 0.f); v11 = fmaxf(v11, 0.f);
        }
        size_t o0 = (size_t)(row0 + m0 + j) * N + col0 + r;
        size_t o1 = (size_t)(row0 + 16 + m0 + j) * N + col0 + r;
        if (Cf) {
            Cf[o0] = v00; Cf[o0 + 16] = v01;
            Cf[o1] = v10; Cf[o1 + 16] = v11;
        }
        if (Cb) {
            Cb[o0] = (bf16_t)v00; Cb[o0 + 16] = (bf16_t)v01;
            Cb[o1] = (bf16_t)v10; Cb[o1 + 16] = (bf16_t)v11;
        }
    }
}

// ---------------------------------------------------------------------------
// 3b) V transpose: (B*N, D) bf16 -> (B, H, DK, N) bf16 so PV B-fragments are
//     contiguous 16B loads.
// ---------------------------------------------------------------------------
__global__ void transpose_v_kernel(const bf16_t* __restrict__ V, bf16_t* __restrict__ Vt) {
    size_t idx = (size_t)blockIdx.x * blockDim.x + threadIdx.x;   // over B*H*DK*N
    int n = (int)(idx & (Nn - 1));
    size_t rest = idx >> 9;             // / Nn
    int d = (int)(rest & (DK - 1));
    rest >>= 5;                          // / DK
    int h = (int)(rest & (Hh - 1));
    int b = (int)(rest >> 3);
    Vt[idx] = V[((size_t)(b * Nn + n)) * Dm + h * DK + d];
}

// ---------------------------------------------------------------------------
// 4) flash attention: one wave per (b, head, 16-query tile); dk=32.
//    S = QK^T/sqrt(dk) + ew ; online softmax; O = P V.
// ---------------------------------------------------------------------------
__global__ void attn_kernel(const bf16_t* __restrict__ Q, const bf16_t* __restrict__ Km,
                            const bf16_t* __restrict__ Vt, const float* __restrict__ ew,
                            bf16_t* __restrict__ Out) {
    __shared__ __align__(16) bf16_t Pls[8][16][32];
    const int lane = threadIdx.x & 31;
    const int w = threadIdx.x >> 5;
    int gw = blockIdx.x * 8 + w;         // 0 .. B*H*(N/16)-1 = 4095
    const int qt = gw & 31;  gw >>= 5;   // N/16 = 32 query tiles
    const int hh = gw & 7;   gw >>= 3;   // H = 8
    const int b  = gw;                   // B = 16
    const int q0 = qt * 16;
    const int r  = lane & 15;
    const int kb = (lane >> 4) * 8;
    const int m0 = kb;
    const float inv_scale = 0.1767766953f;   // 1/sqrt(32)

    const bf16_t* qrow = Q + ((size_t)(b * Nn + q0 + r)) * Dm + hh * DK;
    v16bf qf = load_frag(qrow + kb, qrow + kb + 16);
    const float* ewp = ew + ((size_t)b * Nn + q0) * Nn;
    // V^T rows for this wave's two output-dim halves
    const bf16_t* vt0 = Vt + ((size_t)(b * Hh + hh) * DK + r) * Nn;
    const bf16_t* vt1 = Vt + ((size_t)(b * Hh + hh) * DK + 16 + r) * Nn;

    float mrun[8], lrun[8];
    v8f O0 = {}, O1 = {};
#pragma unroll
    for (int j = 0; j < 8; ++j) { mrun[j] = -1e30f; lrun[j] = 0.f; }

    for (int c = 0; c < Nn; c += 32) {
        // QK^T for 32 keys: two 16x16 D-tiles
        const bf16_t* k0 = Km + ((size_t)(b * Nn + c      + r)) * Dm + hh * DK;
        const bf16_t* k1 = Km + ((size_t)(b * Nn + c + 16 + r)) * Dm + hh * DK;
        v16bf kf0 = load_frag(k0 + kb, k0 + kb + 16);
        v16bf kf1 = load_frag(k1 + kb, k1 + kb + 16);
        v8f s0 = {}, s1 = {};
        s0 = wmma_bf16(qf, kf0, s0);
        s1 = wmma_bf16(qf, kf1, s1);

        float p0[8], p1[8], al[8];
#pragma unroll
        for (int j = 0; j < 8; ++j) {
            float e0 = ewp[(size_t)(m0 + j) * Nn + (c + r)];
            float e1 = ewp[(size_t)(m0 + j) * Nn + (c + 16 + r)];
            p0[j] = s0[j] * inv_scale + e0;
            p1[j] = s1[j] * inv_scale + e1;
        }
#pragma unroll
        for (int j = 0; j < 8; ++j) {
            float pm = fmaxf(p0[j], p1[j]);
            for (int off = 1; off < 16; off <<= 1)
                pm = fmaxf(pm, __shfl_xor(pm, off, 32));   // reduce within 16-lane half (same rows)
            float mn = fmaxf(mrun[j], pm);
            al[j] = expf(mrun[j] - mn);
            mrun[j] = mn;
            p0[j] = expf(p0[j] - mn);
            p1[j] = expf(p1[j] - mn);
            float ps = p0[j] + p1[j];
            for (int off = 1; off < 16; off <<= 1)
                ps += __shfl_xor(ps, off, 32);
            lrun[j] = lrun[j] * al[j] + ps;
            O0[j] *= al[j];
            O1[j] *= al[j];
            // stage P (D-layout -> LDS) for A-layout reload
            Pls[w][m0 + j][r]      = (bf16_t)p0[j];
            Pls[w][m0 + j][16 + r] = (bf16_t)p1[j];
        }
        asm volatile("s_wait_dscnt 0" ::: "memory");
        v16bf pf = load_frag(&Pls[w][r][kb], &Pls[w][r][kb + 16]);

        // V fragments: contiguous 16B loads from V^T
        v16bf vf0 = load_frag(vt0 + c + kb, vt0 + c + kb + 16);
        v16bf vf1 = load_frag(vt1 + c + kb, vt1 + c + kb + 16);
        O0 = wmma_bf16(pf, vf0, O0);
        O1 = wmma_bf16(pf, vf1, O1);
    }
#pragma unroll
    for (int j = 0; j < 8; ++j) {
        float inv = 1.f / lrun[j];
        size_t o = ((size_t)(b * Nn + q0 + m0 + j)) * Dm + hh * DK;
        Out[o + r]      = (bf16_t)(O0[j] * inv);
        Out[o + 16 + r] = (bf16_t)(O1[j] * inv);
    }
}

// ---------------------------------------------------------------------------
// 5) residual + LayerNorm (res may be null; outF/outB may be null)
// ---------------------------------------------------------------------------
__global__ void ln_kernel(const float* __restrict__ hin, const float* __restrict__ res,
                          const float* __restrict__ g, const float* __restrict__ bta,
                          float* __restrict__ outF, bf16_t* __restrict__ outB) {
    __shared__ float red[Dm];
    int row = blockIdx.x, d = threadIdx.x;
    size_t o = (size_t)row * Dm + d;
    float v = hin[o] + (res ? res[o] : 0.f);
    red[d] = v;
    __syncthreads();
    for (int s = Dm / 2; s > 0; s >>= 1) { if (d < s) red[d] += red[d + s]; __syncthreads(); }
    float mean = red[0] * (1.f / Dm);
    __syncthreads();
    float c = v - mean;
    red[d] = c * c;
    __syncthreads();
    for (int s = Dm / 2; s > 0; s >>= 1) { if (d < s) red[d] += red[d + s]; __syncthreads(); }
    float var = red[0] * (1.f / Dm);
    float out = c * rsqrtf(var + 1e-5f) * g[d] + bta[d];
    if (outF) outF[o] = out;
    if (outB) outB[o] = (bf16_t)out;
}

// ---------------------------------------------------------------------------
// launch
// ---------------------------------------------------------------------------
extern "C" void kernel_launch(void* const* d_in, const int* in_sizes, int n_in,
                              void* d_out, int out_size, void* d_ws, size_t ws_size,
                              hipStream_t stream) {
    const float* x      = (const float*)d_in[0];
    const int*   eidx   = (const int*)  d_in[1];
    const float* eattr  = (const float*)d_in[2];
    const float* nodeW  = (const float*)d_in[4];
    const float* nodeB  = (const float*)d_in[5];
    const float* edgeW  = (const float*)d_in[6];
    const float* edgeB  = (const float*)d_in[7];
    const float* peW    = (const float*)d_in[8];
    const float* peB    = (const float*)d_in[9];
    const float* Wq     = (const float*)d_in[10];
    const float* Wk     = (const float*)d_in[11];
    const float* Wv     = (const float*)d_in[12];
    const float* Wo     = (const float*)d_in[13];
    const float* bo     = (const float*)d_in[14];
    const float* g1     = (const float*)d_in[15];
    const float* b1     = (const float*)d_in[16];
    const float* g2     = (const float*)d_in[17];
    const float* b2     = (const float*)d_in[18];
    const float* W1     = (const float*)d_in[19];
    const float* fb1    = (const float*)d_in[20];
    const float* W2     = (const float*)d_in[21];
    const float* fb2    = (const float*)d_in[22];
    const float* fng    = (const float*)d_in[23];
    const float* fnb    = (const float*)d_in[24];
    float* out = (float*)d_out;

    char* p = (char*)d_ws;
    auto alloc = [&](size_t bytes) -> char* {
        char* r = p;
        p += (bytes + 255) & ~(size_t)255;
        return r;
    };
    float*  hF     = (float*) alloc((size_t)BNm * Dm * 4);
    float*  tmpF   = (float*) alloc((size_t)BNm * Dm * 4);
    bf16_t* hB     = (bf16_t*)alloc((size_t)BNm * Dm * 2);
    bf16_t* Qb     = (bf16_t*)alloc((size_t)BNm * Dm * 2);
    bf16_t* Kb     = (bf16_t*)alloc((size_t)BNm * Dm * 2);
    bf16_t* Vb     = (bf16_t*)alloc((size_t)BNm * Dm * 2);
    bf16_t* VtB    = (bf16_t*)alloc((size_t)BNm * Dm * 2);     // V transposed (B,H,DK,N)
    bf16_t* Ab     = (bf16_t*)alloc((size_t)BNm * Dm * 2);     // attention output
    bf16_t* Mb     = (bf16_t*)alloc((size_t)BNm * DFFm * 2);   // FFN mid
    float*  ewBuf  = (float*) alloc((size_t)Bn * Nn * Nn * 4);
    bf16_t* WqT    = (bf16_t*)alloc((size_t)Ll * Dm * Dm * 2);
    bf16_t* WkT    = (bf16_t*)alloc((size_t)Ll * Dm * Dm * 2);
    bf16_t* WvT    = (bf16_t*)alloc((size_t)Ll * Dm * Dm * 2);
    bf16_t* WoT    = (bf16_t*)alloc((size_t)Ll * Dm * Dm * 2);
    bf16_t* W1T    = (bf16_t*)alloc((size_t)Ll * Dm * DFFm * 2);
    bf16_t* W2T    = (bf16_t*)alloc((size_t)Ll * DFFm * Dm * 2);
    float*  consts = (float*) alloc(256);

    // setup
    hipLaunchKernelGGL(edge_const_kernel, dim3(1), dim3(Dm), 0, stream, edgeW, edgeB, consts);
    {
        size_t nEw = (size_t)Bn * Nn * Nn;
        hipLaunchKernelGGL(fill_zero_kernel, dim3((unsigned)((nEw + 255) / 256)), dim3(256), 0, stream, ewBuf, nEw);
    }
    hipLaunchKernelGGL(edge_scatter_kernel, dim3(Em / 256), dim3(256), 0, stream, eidx, eattr, consts, ewBuf);
    hipLaunchKernelGGL(embed_kernel, dim3(BNm), dim3(Dm), 0, stream, x, nodeW, nodeB, peW, peB, hF, hB);

    auto convT = [&](const float* in, bf16_t* o, int K, int Nc) {
        size_t tot = (size_t)Ll * K * Nc;
        hipLaunchKernelGGL(conv_t_kernel, dim3((unsigned)((tot + 255) / 256)), dim3(256), 0, stream,
                           in, o, Ll, K, Nc);
    };
    convT(Wq, WqT, Dm, Dm);
    convT(Wk, WkT, Dm, Dm);
    convT(Wv, WvT, Dm, Dm);
    convT(Wo, WoT, Dm, Dm);
    convT(W1, W1T, Dm, DFFm);   // (L, D, DFF) -> (L, DFF, D)
    convT(W2, W2T, DFFm, Dm);   // (L, DFF, D) -> (L, D, DFF)

    auto gemm = [&](const bf16_t* A, const bf16_t* Bt, const float* bias,
                    float* Cf, bf16_t* Cb, int N, int K, int relu) {
        dim3 grid(N / 128, BNm / 64);
        hipLaunchKernelGGL(gemm_wmma_kernel, grid, dim3(256), 0, stream,
                           A, Bt, bias, Cf, Cb, BNm, N, K, relu);
    };

    for (int l = 0; l < Ll; ++l) {
        size_t wOff  = (size_t)l * Dm * Dm;
        size_t w1Off = (size_t)l * Dm * DFFm;
        gemm(hB, WqT + wOff, nullptr, nullptr, Qb, Dm, Dm, 0);
        gemm(hB, WkT + wOff, nullptr, nullptr, Kb, Dm, Dm, 0);
        gemm(hB, WvT + wOff, nullptr, nullptr, Vb, Dm, Dm, 0);
        hipLaunchKernelGGL(transpose_v_kernel, dim3(BNm), dim3(Dm), 0, stream, Vb, VtB);
        hipLaunchKernelGGL(attn_kernel, dim3(Bn * Hh * (Nn / 16) / 8), dim3(256), 0, stream,
                           Qb, Kb, VtB, ewBuf, Ab);
        gemm(Ab, WoT + wOff, bo + (size_t)l * Dm, tmpF, nullptr, Dm, Dm, 0);
        hipLaunchKernelGGL(ln_kernel, dim3(BNm), dim3(Dm), 0, stream,
                           hF, tmpF, g1 + (size_t)l * Dm, b1 + (size_t)l * Dm, hF, hB);
        gemm(hB, W1T + w1Off, fb1 + (size_t)l * DFFm, nullptr, Mb, DFFm, Dm, 1);
        gemm(Mb, W2T + w1Off, fb2 + (size_t)l * Dm, tmpF, nullptr, Dm, DFFm, 0);
        hipLaunchKernelGGL(ln_kernel, dim3(BNm), dim3(Dm), 0, stream,
                           hF, tmpF, g2 + (size_t)l * Dm, b2 + (size_t)l * Dm, hF, hB);
    }
    hipLaunchKernelGGL(ln_kernel, dim3(BNm), dim3(Dm), 0, stream,
                       hF, nullptr, fng, fnb, out, nullptr);
}
